// LshSelfAttention_84344567759092
// MI455X (gfx1250) — matrix operand order repaired and checked by reading, not compiled, not run on hardware
//
#include <hip/hip_runtime.h>

// ---------------- problem constants (from reference) ----------------
#define L_SEQ    4096
#define HID_DIM  1024
#define NHEAD    16
#define DHEAD    64
#define NBATCH   2
#define NROUNDS  2              // NH hash rounds
#define NBUCKETS 64
#define BS_CHUNK 64
#define NBH      32             // NBATCH*NHEAD
#define SLEN     8192           // NROUNDS*L_SEQ (sorted length per n)
#define NCHUNK   128            // SLEN / BS_CHUNK

typedef __bf16 bf16;
typedef __attribute__((ext_vector_type(16))) __bf16 bf16x16;
typedef __attribute__((ext_vector_type(8)))  __bf16 bf16x8;
typedef __attribute__((ext_vector_type(8)))  float  f32x8;
typedef __attribute__((ext_vector_type(4)))  float  f32x4;

union BF16Frag { bf16x16 v; bf16x8 h[2]; __bf16 e[16]; };

// ---------------- WMMA fragment helpers (wave32 layouts, CDNA5 ISA 7.12.2) ----
// A fragment (16x32 bf16, MxK), A row-major [M][K]: lane m=lane&15, hs=lane>>4.
// elements 0..7  <- K = hs*8 + 0..7 ; elements 8..15 <- K = 16 + hs*8 + 0..7
__device__ __forceinline__ bf16x16 load_afrag(const __bf16* A, int lda, int mBase, int kBase) {
  int lane = threadIdx.x & 31;
  int hs = lane >> 4, m = lane & 15;
  const __bf16* p = A + (size_t)(mBase + m) * (size_t)lda + kBase;
  BF16Frag f;
  f.h[0] = *(const bf16x8*)(p + hs * 8);
  f.h[1] = *(const bf16x8*)(p + 16 + hs * 8);
  return f.v;
}

// Same A layout but fp32 source with on-the-fly bf16 convert (for LDS-resident P / global Q).
__device__ __forceinline__ bf16x16 cvt_rowf32_afrag(const float* row, int hs) {
  f32x4 a0 = *(const f32x4*)(row + hs * 8);
  f32x4 a1 = *(const f32x4*)(row + hs * 8 + 4);
  f32x4 b0 = *(const f32x4*)(row + 16 + hs * 8);
  f32x4 b1 = *(const f32x4*)(row + 16 + hs * 8 + 4);
  BF16Frag f;
#pragma unroll
  for (int i = 0; i < 4; ++i) {
    f.e[i]      = (__bf16)a0[i];
    f.e[4 + i]  = (__bf16)a1[i];
    f.e[8 + i]  = (__bf16)b0[i];
    f.e[12 + i] = (__bf16)b1[i];
  }
  return f.v;
}

// B fragment (32x16 bf16, KxN) from a matrix stored as BT[N][K] (row-major):
// lane n=lane&15 picks the column, hs=lane>>4; elements 0..7 <- K = hs*8 + 0..7 ; 8..15 <- K = 16 + hs*8 + 0..7.
__device__ __forceinline__ bf16x16 load_bfrag(const __bf16* BT, int ldb, int nBase, int kBase) {
  int lane = threadIdx.x & 31;
  int hs = lane >> 4, nn = lane & 15;
  const __bf16* p = BT + (size_t)(nBase + nn) * (size_t)ldb + kBase;
  BF16Frag f;
  f.h[0] = *(const bf16x8*)(p + hs * 8);
  f.h[1] = *(const bf16x8*)(p + 16 + hs * 8);
  return f.v;
}

__device__ __forceinline__ f32x8 wmma_bf16(bf16x16 a, bf16x16 b, f32x8 c) {
  // v_wmma_f32_16x16x32_bf16 : D = A*B + C, fp32 accumulate
  return __builtin_amdgcn_wmma_f32_16x16x32_bf16(false, a, false, b, (short)0, c, false, false);
}

// ---------------- elementwise fp32 -> bf16 convert ----------------
__global__ void cvt_bf16_kernel(const float* __restrict__ in, __bf16* __restrict__ out, int n) {
  int idx = blockIdx.x * blockDim.x + threadIdx.x;
  if (idx < n) out[idx] = (__bf16)in[idx];
}

// ---------------- weight transpose + bf16 convert (1024x1024) ----------------
// Wqk [e][h*d] -> WqkT [h*d][e];  Wo [h*d][e] -> WoT [e][h*d] : both plain transposes.
__global__ void transpose_cvt_kernel(const float* __restrict__ in, __bf16* __restrict__ out) {
  int idx = blockIdx.x * blockDim.x + threadIdx.x;  // r*1024 + c
  int r = idx >> 10, cc = idx & 1023;
  out[idx] = (__bf16)in[(size_t)cc * 1024 + r];
}

// ---------------- GEMM: C(f32)[8192][1024] = A(bf16)[8192][1024] x BT(bf16)[1024][1024] --
// 8 waves per block (4x2); each wave computes 32x64 (2x4 WMMA tiles): 8 WMMA / 6 frag loads.
// Dims hardcoded (all three GEMMs are 8192x1024x1024) so the k-loop fully pipelines.
// mode 0: C row-major [M][N].  mode 1: scatter [b,l][h,d] -> [b,h,l,d].
__global__ __launch_bounds__(256)
void gemm_bf16(const __bf16* __restrict__ A, const __bf16* __restrict__ BT,
               float* __restrict__ C, int mode) {
  const int K = HID_DIM;
  int wave = threadIdx.x >> 5;
  int lane = threadIdx.x & 31;
  int wy = wave >> 1, wx = wave & 1;
  int mBase = blockIdx.y * 128 + wy * 32;
  int nBase = blockIdx.x * 128 + wx * 64;
  int m = lane & 15, hs = lane >> 4;
  f32x8 acc[2][4] = {};
#pragma unroll 2
  for (int k = 0; k < K; k += 32) {
    // prefetch two k-steps ahead (global_prefetch_b8; speculative, OOB dropped)
    __builtin_prefetch(A + (size_t)(mBase + m) * K + k + 64, 0, 1);
    __builtin_prefetch(BT + (size_t)(nBase + (lane & 63)) * K + k + 64, 0, 1);
    bf16x16 a0 = load_afrag(A, K, mBase, k);
    bf16x16 a1 = load_afrag(A, K, mBase + 16, k);
    bf16x16 b0 = load_bfrag(BT, K, nBase, k);
    bf16x16 b1 = load_bfrag(BT, K, nBase + 16, k);
    bf16x16 b2 = load_bfrag(BT, K, nBase + 32, k);
    bf16x16 b3 = load_bfrag(BT, K, nBase + 48, k);
    acc[0][0] = wmma_bf16(a0, b0, acc[0][0]);
    acc[0][1] = wmma_bf16(a0, b1, acc[0][1]);
    acc[0][2] = wmma_bf16(a0, b2, acc[0][2]);
    acc[0][3] = wmma_bf16(a0, b3, acc[0][3]);
    acc[1][0] = wmma_bf16(a1, b0, acc[1][0]);
    acc[1][1] = wmma_bf16(a1, b1, acc[1][1]);
    acc[1][2] = wmma_bf16(a1, b2, acc[1][2]);
    acc[1][3] = wmma_bf16(a1, b3, acc[1][3]);
  }
#pragma unroll
  for (int i = 0; i < 2; ++i)
#pragma unroll
    for (int j = 0; j < 4; ++j) {
      int col = nBase + j * 16 + m;
#pragma unroll
      for (int kk = 0; kk < 8; ++kk) {
        int row = mBase + i * 16 + hs * 8 + kk;
        size_t idx;
        if (mode == 0) {
          idx = (size_t)row * HID_DIM + col;
        } else {
          // row = b*L + l ; col = head*64 + d  ->  [(b*H+head)*L + l]*64 + d
          idx = (size_t)((row >> 12) * NHEAD + (col >> 6)) * ((size_t)L_SEQ * DHEAD)
              + (size_t)(row & (L_SEQ - 1)) * DHEAD + (col & (DHEAD - 1));
        }
        C[idx] = acc[i][j][kk];
      }
    }
}

// ---------------- LSH bucket assignment ----------------
// key = (bucket + h*NBUCKETS)*L + pos ; argmax over [rot, -rot] with first-tie.
__global__ void lsh_hash_kernel(const float* __restrict__ qk,
                                const float* __restrict__ rot,
                                int* __restrict__ keys) {
  int idx = blockIdx.x * blockDim.x + threadIdx.x;  // n*L + l
  if (idx >= NBH * L_SEQ) return;
  int n = idx >> 12, l = idx & (L_SEQ - 1);
  const float* qr = qk + (size_t)idx * DHEAD;
#pragma unroll
  for (int h = 0; h < NROUNDS; ++h) {
    float dots[32];
#pragma unroll
    for (int r = 0; r < 32; ++r) dots[r] = 0.f;
    for (int d = 0; d < DHEAD; ++d) {
      float qd = qr[d];
      const float* rr = rot + ((size_t)d * NROUNDS + h) * 32;  // rot[d][h][r]
#pragma unroll
      for (int r = 0; r < 32; ++r) dots[r] += qd * rr[r];
    }
    float best = dots[0];
    int bb = 0;
#pragma unroll
    for (int r = 1; r < 32; ++r) if (dots[r] > best) { best = dots[r]; bb = r; }
#pragma unroll
    for (int r = 0; r < 32; ++r) if (-dots[r] > best) { best = -dots[r]; bb = 32 + r; }
    keys[(size_t)n * SLEN + h * L_SEQ + l] = ((bb + h * NBUCKETS) << 12) | l;
  }
}

// ---------------- stable counting sort (128 buckets) ----------------
__global__ __launch_bounds__(128)
void lsh_sort_kernel(const int* __restrict__ keys, int* __restrict__ sticker,
                     int* __restrict__ undo) {
  __shared__ unsigned char buck[SLEN];
  __shared__ int cnt[128];
  __shared__ int start[128];
  int n = blockIdx.x, t = threadIdx.x;
  const int* kn = keys + (size_t)n * SLEN;
  for (int j = t; j < SLEN; j += 128) buck[j] = (unsigned char)(kn[j] >> 12);
  __syncthreads();
  int c = 0;
  for (int j = 0; j < SLEN; ++j) c += (buck[j] == t);
  cnt[t] = c;
  __syncthreads();
  if (t == 0) { int s = 0; for (int b = 0; b < 128; ++b) { start[b] = s; s += cnt[b]; } }
  __syncthreads();
  int off = start[t];
  int* stn = sticker + (size_t)n * SLEN;
  int* un  = undo + (size_t)n * SLEN;
  for (int j = 0; j < SLEN; ++j)
    if (buck[j] == t) { stn[off] = j; un[j] = off; ++off; }
}

// ---------------- chunked local attention (WMMA) ----------------
// One block = one (n, chunk). 64 queries attend to 128 kv (chunk + prev chunk).
__global__ __launch_bounds__(256)
void lsh_attn_kernel(const float* __restrict__ qk, const float* __restrict__ v,
                     const unsigned char* __restrict__ pad,
                     const int* __restrict__ sticker,
                     float* __restrict__ so, float* __restrict__ slog) {
  __shared__ __align__(16) float S[BS_CHUNK][2 * BS_CHUNK];        // scores -> probs (32 KB)
  __shared__ __align__(16) __bf16 KV[2 * BS_CHUNK * DHEAD];        // Kn then V^T (16 KB, reused)
  __shared__ int qpos[BS_CHUNK];
  __shared__ int kpos[2 * BS_CHUNK];
  __shared__ int kpad[2 * BS_CHUNK];

  int n = blockIdx.y, c = blockIdx.x;
  int t = threadIdx.x;
  int lane = t & 31, wave = t >> 5;
  int m = lane & 15, hs = lane >> 4;
  int cprev = (c + NCHUNK - 1) & (NCHUNK - 1);
  const int* stn = sticker + (size_t)n * SLEN;

  if (t < BS_CHUNK) qpos[t] = stn[c * BS_CHUNK + t] & (L_SEQ - 1);
  if (t < 2 * BS_CHUNK) {
    int g = (t < BS_CHUNK) ? (c * BS_CHUNK + t) : (cprev * BS_CHUNK + (t - BS_CHUNK));
    int tok = stn[g] & (L_SEQ - 1);
    kpos[t] = tok;
    kpad[t] = pad[(size_t)(n >> 4) * L_SEQ + tok];
  }
  __syncthreads();

  // Stage normalized keys Kn[j][d] into KV (bk = bqk * rsqrt(max(|bqk|^2,1e-12)))
  {
    int j = t >> 1, s = t & 1;  // 2 lanes per row
    const float* src = qk + ((size_t)n * L_SEQ + kpos[j]) * DHEAD;
    float ss = 0.f;
    for (int d = s * 32; d < s * 32 + 32; ++d) { float x0 = src[d]; ss += x0 * x0; }
    ss += __shfl_xor(ss, 1, 32);
    float inv = rsqrtf(fmaxf(ss, 1e-12f));
    for (int d = s * 32; d < s * 32 + 32; ++d) KV[j * DHEAD + d] = (__bf16)(src[d] * inv);
  }
  __syncthreads();

  // Scores: S = Q * Kn^T. 8 waves cover 4 i-tiles x 8 j-tiles, K-dim = 64 (2 steps).
  int rt = wave & 3;      // i-tile
  int cg = wave >> 2;     // group of 4 j-tiles
  f32x8 acc[4] = {};
  const float* qrow = qk + ((size_t)n * L_SEQ + qpos[rt * 16 + m]) * DHEAD;
#pragma unroll
  for (int ks = 0; ks < DHEAD; ks += 32) {
    bf16x16 a = cvt_rowf32_afrag(qrow + ks, hs);
#pragma unroll
    for (int cc = 0; cc < 4; ++cc) {
      bf16x16 b = load_bfrag(KV, DHEAD, (cg * 4 + cc) * 16, ks);  // Kn is [j][d] = [N][K]
      acc[cc] = wmma_bf16(a, b, acc[cc]);
    }
  }
#pragma unroll
  for (int cc = 0; cc < 4; ++cc) {
    int col = (cg * 4 + cc) * 16 + m;
#pragma unroll
    for (int kk = 0; kk < 8; ++kk) S[rt * 16 + hs * 8 + kk][col] = acc[cc][kk];
  }
  __syncthreads();  // all waves done reading Kn -> safe to overwrite KV

  // Stage V^T[d][j] into KV (B-side [N][K] layout for the P*V GEMM)
  for (int e = t; e < 2 * BS_CHUNK * DHEAD; e += 256) {
    int d = e >> 7, j = e & (2 * BS_CHUNK - 1);
    KV[d * (2 * BS_CHUNK) + j] = (__bf16)v[((size_t)n * L_SEQ + kpos[j]) * DHEAD + d];
  }
  __syncthreads();

  // Masks + row-wise logsumexp softmax (4 lanes per row, shfl reductions)
  {
    int r = t >> 2, s = t & 3;
    int qp = qpos[r];
    float mx = -3.0e38f;
    for (int j = s * 32; j < s * 32 + 32; ++j) {
      float sv_ = S[r][j] * 0.125f;        // D^-0.5
      if (qp == kpos[j]) sv_ += -1.0e5f;   // shared-QK self mask
      if (kpad[j]) sv_ = -1.0e9f;          // padding mask
      S[r][j] = sv_;
      mx = fmaxf(mx, sv_);
    }
    mx = fmaxf(mx, __shfl_xor(mx, 1, 32));
    mx = fmaxf(mx, __shfl_xor(mx, 2, 32));
    float sum = 0.f;
    for (int j = s * 32; j < s * 32 + 32; ++j) sum += __expf(S[r][j] - mx);
    sum += __shfl_xor(sum, 1, 32);
    sum += __shfl_xor(sum, 2, 32);
    float lse = mx + __logf(sum);
    for (int j = s * 32; j < s * 32 + 32; ++j) S[r][j] = __expf(S[r][j] - lse);
    if (s == 0) slog[(size_t)n * SLEN + c * BS_CHUNK + r] = lse;
  }
  __syncthreads();

  // O = P * V : 4 i-tiles x 4 d-tiles, K-dim = 128 (4 steps). P f32->bf16 on load.
  int it = wave & 3, dg = wave >> 2;
  f32x8 oacc[2] = {};
  const float* prow = &S[it * 16 + m][0];
#pragma unroll
  for (int ks = 0; ks < 2 * BS_CHUNK; ks += 32) {
    bf16x16 a = cvt_rowf32_afrag(prow + ks, hs);
#pragma unroll
    for (int dd = 0; dd < 2; ++dd) {
      bf16x16 b = load_bfrag(KV, 2 * BS_CHUNK, (dg * 2 + dd) * 16, ks);  // V^T [d][j]
      oacc[dd] = wmma_bf16(a, b, oacc[dd]);
    }
  }
#pragma unroll
  for (int dd = 0; dd < 2; ++dd) {
    int dcol = (dg * 2 + dd) * 16 + m;
#pragma unroll
    for (int kk = 0; kk < 8; ++kk) {
      int r = it * 16 + hs * 8 + kk;
      so[((size_t)n * SLEN + c * BS_CHUNK + r) * DHEAD + dcol] = oacc[dd][kk];
    }
  }
}

// ---------------- unsort + combine hash rounds (softmax over NH logits) -------
// Writes the combined context directly as bf16 in [b*l][head*64+d] layout (A of out-proj).
__global__ void lsh_combine_kernel(const float* __restrict__ so, const float* __restrict__ slog,
                                   const int* __restrict__ undo, __bf16* __restrict__ att) {
  int idx = blockIdx.x * blockDim.x + threadIdx.x;  // n*L + l
  if (idx >= NBH * L_SEQ) return;
  int n = idx >> 12, l = idx & (L_SEQ - 1);
  const int* un = undo + (size_t)n * SLEN;
  int i0 = un[l];            // round 0: sorted position of (h=0, l)
  int i1 = un[L_SEQ + l];    // round 1
  float l0 = slog[(size_t)n * SLEN + i0];
  float l1 = slog[(size_t)n * SLEN + i1];
  float mx = fmaxf(l0, l1);
  float e0 = __expf(l0 - mx), e1 = __expf(l1 - mx);
  float inv = 1.f / (e0 + e1);
  float w0 = e0 * inv, w1 = e1 * inv;
  const float* o0 = so + ((size_t)n * SLEN + i0) * DHEAD;
  const float* o1 = so + ((size_t)n * SLEN + i1) * DHEAD;
  int b = n >> 4, head = n & 15;
  __bf16* dst = att + ((size_t)(b * L_SEQ + l)) * HID_DIM + head * DHEAD;
  for (int d = 0; d < DHEAD; ++d) dst[d] = (__bf16)(w0 * o0[d] + w1 * o1[d]);
}

// ---------------- launch ----------------
extern "C" void kernel_launch(void* const* d_in, const int* in_sizes, int n_in,
                              void* d_out, int out_size, void* d_ws, size_t ws_size,
                              hipStream_t stream) {
  (void)in_sizes; (void)n_in; (void)out_size; (void)ws_size;
  const float* x            = (const float*)d_in[0];
  const unsigned char* pad  = (const unsigned char*)d_in[1];
  const float* Wqk          = (const float*)d_in[2];
  const float* Wv           = (const float*)d_in[3];
  const float* Wo           = (const float*)d_in[4];
  const float* rot          = (const float*)d_in[5];
  float* out = (float*)d_out;

  char* ws = (char*)d_ws;
  size_t off = 0;
  auto alloc = [&](size_t bytes) -> void* {
    void* p = ws + off;
    off += (bytes + 255) & ~(size_t)255;
    return p;
  };
  __bf16* WqkT  = (__bf16*)alloc((size_t)HID_DIM * HID_DIM * sizeof(__bf16));
  __bf16* WvT   = (__bf16*)alloc((size_t)HID_DIM * HID_DIM * sizeof(__bf16));
  __bf16* WoT   = (__bf16*)alloc((size_t)HID_DIM * HID_DIM * sizeof(__bf16));
  __bf16* xbf   = (__bf16*)alloc((size_t)NBATCH * L_SEQ * HID_DIM * sizeof(__bf16));
  float*  qkbuf = (float*)alloc((size_t)NBH * L_SEQ * DHEAD * sizeof(float));   // [n][l][d]
  float*  vbuf  = (float*)alloc((size_t)NBH * L_SEQ * DHEAD * sizeof(float));
  int*    keys  = (int*)alloc((size_t)NBH * SLEN * sizeof(int));
  int*    stick = (int*)alloc((size_t)NBH * SLEN * sizeof(int));
  int*    undo  = (int*)alloc((size_t)NBH * SLEN * sizeof(int));
  float*  slogb = (float*)alloc((size_t)NBH * SLEN * sizeof(float));
  float*  sob   = (float*)alloc((size_t)NBH * SLEN * DHEAD * sizeof(float));
  __bf16* attb  = (__bf16*)alloc((size_t)NBATCH * L_SEQ * HID_DIM * sizeof(__bf16));

  // 1) bf16 operand prep: weight transposes + activation convert
  transpose_cvt_kernel<<<4096, 256, 0, stream>>>(Wqk, WqkT);
  transpose_cvt_kernel<<<4096, 256, 0, stream>>>(Wv,  WvT);
  transpose_cvt_kernel<<<4096, 256, 0, stream>>>(Wo,  WoT);
  cvt_bf16_kernel<<<(NBATCH * L_SEQ * HID_DIM) / 256, 256, 0, stream>>>(
      x, xbf, NBATCH * L_SEQ * HID_DIM);

  // 2) shared-QK and V projections, scattered straight into [b,h,l,d]
  dim3 ggrid(HID_DIM / 128, (NBATCH * L_SEQ) / 128);
  gemm_bf16<<<ggrid, 256, 0, stream>>>(xbf, WqkT, qkbuf, 1);
  gemm_bf16<<<ggrid, 256, 0, stream>>>(xbf, WvT,  vbuf,  1);

  // 3) LSH bucket keys; 4) stable counting sort per (b,h)
  lsh_hash_kernel<<<(NBH * L_SEQ) / 256, 256, 0, stream>>>(qkbuf, rot, keys);
  lsh_sort_kernel<<<NBH, 128, 0, stream>>>(keys, stick, undo);

  // 5) chunked local attention over sorted tokens (WMMA bf16, fp32 accum)
  lsh_attn_kernel<<<dim3(NCHUNK, NBH), 256, 0, stream>>>(qkbuf, vbuf, pad, stick, sob, slogb);

  // 6) unsort + combine the two hash rounds with logsumexp weights (emit bf16)
  lsh_combine_kernel<<<(NBH * L_SEQ) / 256, 256, 0, stream>>>(sob, slogb, undo, attb);

  // 7) output projection
  gemm_bf16<<<ggrid, 256, 0, stream>>>(attb, WoT, out, 0);
}
